// EventAwareAttention_36438502540095
// MI455X (gfx1250) — matrix operand order repaired and checked by reading, not compiled, not run on hardware
//
#include <hip/hip_runtime.h>

// Problem constants (from reference): B=8, S=2048, H=512, DK=512
#define B_ 8
#define S_ 2048
#define H_ 512

typedef __bf16 bf16_t;
typedef __attribute__((ext_vector_type(16))) __bf16 v16bf;
typedef __attribute__((ext_vector_type(8)))  __bf16 v8bf;
typedef __attribute__((ext_vector_type(4)))  __bf16 v4bf;
typedef __attribute__((ext_vector_type(8)))  float  v8f;
typedef __attribute__((ext_vector_type(4)))  float  v4f;

// ---------------- CDNA5 async global->LDS copy (ASYNCcnt-tracked) ----------
// ROCm 7.2's __builtin_amdgcn_global_load_async_to_lds_b128 has a non-upstream
// signature (v4i* first arg), so issue the instruction via inline asm per the
// VGLOBAL encoding: global_load_async_to_lds_b128 vdst(LDS addr), vaddr64, off
#define USE_ASYNC_COPY 1

#if USE_ASYNC_COPY
__device__ __forceinline__ void async_copy_b128(const bf16_t* g, bf16_t* l) {
    const unsigned lofs =
        (unsigned)(unsigned long long)(__attribute__((address_space(3))) char*)(char*)l;
    asm volatile("global_load_async_to_lds_b128 %0, %1, off"
                 :
                 : "v"(lofs), "v"(g)
                 : "memory");
}
template <int N>
__device__ __forceinline__ void wait_async() {
#if __has_builtin(__builtin_amdgcn_s_wait_asynccnt)
    __builtin_amdgcn_s_wait_asynccnt((unsigned short)N);
#else
    asm volatile("s_wait_asynccnt %0" ::"i"(N) : "memory");
#endif
}
#endif

// ---- WMMA fragment loaders (layouts per CDNA5 ISA 7.12.2, wave32) ----
// LDS A tile layout: [rows][LDK k] bf16 (k contiguous per row)
// A 16x32 bf16 fragment: lanes 0-15: M=lane, elems = K0..7,K16..23
//                        lanes 16-31: M=lane-16, elems = K8..15,K24..31
template <int LDK>
__device__ __forceinline__ v16bf ldsA_frag(const bf16_t* A, int lane, int mtile, int kc) {
    const int row = mtile + (lane & 15);
    const int c0  = ((lane < 16) ? 0 : 8) + kc * 32;
    union { v16bf v; v8bf h[2]; } u;
    u.h[0] = *(const v8bf*)(A + row * LDK + c0);
    u.h[1] = *(const v8bf*)(A + row * LDK + c0 + 16);
    return u.v;
}
// B 32x16 bf16 fragment: lanes 0-15: N=lane, K0..7,K16..23 ; lanes 16-31: N=lane-16, K8..15,K24..31
template <int LDK>
__device__ __forceinline__ v16bf ldsB_frag(const bf16_t* Bm, int lane, int ntile, int kc) {
    const int row = ntile + (lane & 15);
    const int k0  = ((lane < 16) ? 0 : 8) + kc * 32;
    union { v16bf v; v8bf h[2]; } u;
    u.h[0] = *(const v8bf*)(Bm + row * LDK + k0);
    u.h[1] = *(const v8bf*)(Bm + row * LDK + k0 + 16);
    return u.v;
}

__device__ __forceinline__ v8f wmma_bf16(v16bf a, v16bf b, v8f c) {
    return __builtin_amdgcn_wmma_f32_16x16x32_bf16(false, a, false, b, (short)0, c, false, false);
}

// =====================================================================
// Kernel 1: QKV projection.  Out[m,n] = sum_h X[m,h]*W[n,h] + bias[n]
// WG tile 128(M) x 64(N), 8 waves as 4x2, wave tile 32x32 (4 WMMA/step).
// f32 sources -> bf16 via VGPR staging; double-buffered LDS, 1 barrier/iter.
// =====================================================================
__global__ __launch_bounds__(256) void qkv_kernel(const float* __restrict__ X,
                                                  const float* __restrict__ W,
                                                  const float* __restrict__ bias,
                                                  bf16_t* __restrict__ Out) {
    __shared__ __align__(16) bf16_t Asm[2][128 * 32];
    __shared__ __align__(16) bf16_t Bsm[2][64 * 32];
    const int tid = threadIdx.x, lane = tid & 31, wid = tid >> 5;
    const int m0 = blockIdx.x * 128, n0 = blockIdx.y * 64;
    const int mw = (wid & 3) * 32, nw = (wid >> 2) * 32;
    const int NIT = H_ / 32;

    v4f aReg[4], bReg[2];
    auto loadG = [&](int kt) {
        const int k0 = kt * 32;
#pragma unroll
        for (int j = 0; j < 4; ++j) {  // A: 128x32 f32 = 1024 float4
            const int idx = tid + j * 256, row = idx >> 3, c4 = idx & 7;
            aReg[j] = *(const v4f*)(X + (size_t)(m0 + row) * H_ + k0 + c4 * 4);
        }
#pragma unroll
        for (int j = 0; j < 2; ++j) {  // B: 64x32 f32 = 512 float4
            const int idx = tid + j * 256, row = idx >> 3, c4 = idx & 7;
            bReg[j] = *(const v4f*)(W + (size_t)(n0 + row) * H_ + k0 + c4 * 4);
        }
    };
    auto storeL = [&](int buf) {
#pragma unroll
        for (int j = 0; j < 4; ++j) {
            const int idx = tid + j * 256, row = idx >> 3, c4 = idx & 7;
            v4bf t;
            t[0] = (bf16_t)aReg[j][0]; t[1] = (bf16_t)aReg[j][1];
            t[2] = (bf16_t)aReg[j][2]; t[3] = (bf16_t)aReg[j][3];
            *(v4bf*)(Asm[buf] + row * 32 + c4 * 4) = t;
        }
#pragma unroll
        for (int j = 0; j < 2; ++j) {
            const int idx = tid + j * 256, row = idx >> 3, c4 = idx & 7;
            v4bf t;
            t[0] = (bf16_t)bReg[j][0]; t[1] = (bf16_t)bReg[j][1];
            t[2] = (bf16_t)bReg[j][2]; t[3] = (bf16_t)bReg[j][3];
            *(v4bf*)(Bsm[buf] + row * 32 + c4 * 4) = t;
        }
    };

    v8f c00 = {}, c01 = {}, c10 = {}, c11 = {};
    loadG(0);
    storeL(0);
    for (int kt = 0; kt < NIT; ++kt) {
        if (kt + 1 < NIT) loadG(kt + 1);   // prefetch overlaps WMMA below
        __syncthreads();                   // tile kt visible in LDS
        const bf16_t* A_ = Asm[kt & 1];
        const bf16_t* Bm_ = Bsm[kt & 1];
        v16bf a0 = ldsA_frag<32>(A_, lane, mw, 0);
        v16bf a1 = ldsA_frag<32>(A_, lane, mw + 16, 0);
        v16bf b0 = ldsB_frag<32>(Bm_, lane, nw, 0);
        v16bf b1 = ldsB_frag<32>(Bm_, lane, nw + 16, 0);
        c00 = wmma_bf16(a0, b0, c00); c01 = wmma_bf16(a0, b1, c01);
        c10 = wmma_bf16(a1, b0, c10); c11 = wmma_bf16(a1, b1, c11);
        if (kt + 1 < NIT) storeL((kt + 1) & 1);
    }
    const int nLo = n0 + nw + (lane & 15), nHi = nLo + 16;
    const float bv0 = bias[nLo], bv1 = bias[nHi];
    const int mbase = m0 + mw + ((lane < 16) ? 0 : 8);
#pragma unroll
    for (int r = 0; r < 8; ++r) {
        Out[(size_t)(mbase + r) * H_ + nLo]      = (bf16_t)(c00[r] + bv0);
        Out[(size_t)(mbase + r) * H_ + nHi]      = (bf16_t)(c01[r] + bv1);
        Out[(size_t)(mbase + 16 + r) * H_ + nLo] = (bf16_t)(c10[r] + bv0);
        Out[(size_t)(mbase + 16 + r) * H_ + nHi] = (bf16_t)(c11[r] + bv1);
    }
}

// =====================================================================
// Kernel 2: scores[b,q,k] = (Q.K^T)*shape_ratio/sqrt(512) + ew*flag[b,q]
// WG tile 128(q) x 64(key), K-tile 64 (8 WMMA/wave/barrier).
// bf16 sources: async global->LDS copies (ASYNCcnt), triple-buffer ring.
// =====================================================================
__global__ __launch_bounds__(256) void scores_kernel(const bf16_t* __restrict__ Q,
                                                     const bf16_t* __restrict__ K,
                                                     const float* __restrict__ event_flag,
                                                     const float* __restrict__ event_weight,
                                                     const float* __restrict__ shape_ratio,
                                                     float* __restrict__ attn) {
#if USE_ASYNC_COPY
    __shared__ __align__(16) bf16_t Asm[3][128 * 64];
    __shared__ __align__(16) bf16_t Bsm[3][64 * 64];
#else
    __shared__ __align__(16) bf16_t Asm[2][128 * 64];
    __shared__ __align__(16) bf16_t Bsm[2][64 * 64];
#endif
    const int tid = threadIdx.x, lane = tid & 31, wid = tid >> 5;
    const int b = blockIdx.z;
    const int m0 = blockIdx.x * 128, n0 = blockIdx.y * 64;
    const bf16_t* Qb = Q + (size_t)b * S_ * H_;
    const bf16_t* Kb = K + (size_t)b * S_ * H_;
    const int mw = (wid & 3) * 32, nw = (wid >> 2) * 32;
    const int NIT = H_ / 64;  // 8
    v8f c00 = {}, c01 = {}, c10 = {}, c11 = {};

#if USE_ASYNC_COPY
    // 6 async b128 per thread per tile: A 128x64 (4), B 64x64 (2)
    auto issue = [&](int kt, int buf) {
        const int k0 = kt * 64;
#pragma unroll
        for (int j = 0; j < 4; ++j) {
            const int idx = tid + j * 256, row = idx >> 3, c8 = idx & 7;
            async_copy_b128(Qb + (size_t)(m0 + row) * H_ + k0 + c8 * 8,
                            Asm[buf] + row * 64 + c8 * 8);
        }
#pragma unroll
        for (int j = 0; j < 2; ++j) {
            const int idx = tid + j * 256, row = idx >> 3, c8 = idx & 7;
            async_copy_b128(Kb + (size_t)(n0 + row) * H_ + k0 + c8 * 8,
                            Bsm[buf] + row * 64 + c8 * 8);
        }
    };
    issue(0, 0);
    for (int kt = 0; kt < NIT; ++kt) {
        if (kt + 1 < NIT) {
            issue(kt + 1, (kt + 1) % 3);  // ring-of-3: target last read at kt-2
            wait_async<6>();              // tile kt complete, kt+1 in flight
        } else {
            wait_async<0>();
        }
        __syncthreads();
        const bf16_t* A_ = Asm[kt % 3];
        const bf16_t* Bm_ = Bsm[kt % 3];
#pragma unroll
        for (int kc = 0; kc < 2; ++kc) {
            v16bf a0 = ldsA_frag<64>(A_, lane, mw, kc);
            v16bf a1 = ldsA_frag<64>(A_, lane, mw + 16, kc);
            v16bf b0 = ldsB_frag<64>(Bm_, lane, nw, kc);
            v16bf b1 = ldsB_frag<64>(Bm_, lane, nw + 16, kc);
            c00 = wmma_bf16(a0, b0, c00); c01 = wmma_bf16(a0, b1, c01);
            c10 = wmma_bf16(a1, b0, c10); c11 = wmma_bf16(a1, b1, c11);
        }
    }
#else
    v8bf aReg[4], bReg[2];
    auto loadG = [&](int kt) {
        const int k0 = kt * 64;
#pragma unroll
        for (int j = 0; j < 4; ++j) {
            const int idx = tid + j * 256, row = idx >> 3, c8 = idx & 7;
            aReg[j] = *(const v8bf*)(Qb + (size_t)(m0 + row) * H_ + k0 + c8 * 8);
        }
#pragma unroll
        for (int j = 0; j < 2; ++j) {
            const int idx = tid + j * 256, row = idx >> 3, c8 = idx & 7;
            bReg[j] = *(const v8bf*)(Kb + (size_t)(n0 + row) * H_ + k0 + c8 * 8);
        }
    };
    auto storeL = [&](int buf) {
#pragma unroll
        for (int j = 0; j < 4; ++j) {
            const int idx = tid + j * 256, row = idx >> 3, c8 = idx & 7;
            *(v8bf*)(Asm[buf] + row * 64 + c8 * 8) = aReg[j];
        }
#pragma unroll
        for (int j = 0; j < 2; ++j) {
            const int idx = tid + j * 256, row = idx >> 3, c8 = idx & 7;
            *(v8bf*)(Bsm[buf] + row * 64 + c8 * 8) = bReg[j];
        }
    };
    loadG(0);
    storeL(0);
    for (int kt = 0; kt < NIT; ++kt) {
        if (kt + 1 < NIT) loadG(kt + 1);
        __syncthreads();
        const bf16_t* A_ = Asm[kt & 1];
        const bf16_t* Bm_ = Bsm[kt & 1];
#pragma unroll
        for (int kc = 0; kc < 2; ++kc) {
            v16bf a0 = ldsA_frag<64>(A_, lane, mw, kc);
            v16bf a1 = ldsA_frag<64>(A_, lane, mw + 16, kc);
            v16bf b0 = ldsB_frag<64>(Bm_, lane, nw, kc);
            v16bf b1 = ldsB_frag<64>(Bm_, lane, nw + 16, kc);
            c00 = wmma_bf16(a0, b0, c00); c01 = wmma_bf16(a0, b1, c01);
            c10 = wmma_bf16(a1, b0, c10); c11 = wmma_bf16(a1, b1, c11);
        }
        if (kt + 1 < NIT) storeL((kt + 1) & 1);
    }
#endif
    const float scale = shape_ratio[0] * 0.044194173824159216f;  // 1/sqrt(512)
    const float ew = event_weight[0];
    const int nLo = n0 + nw + (lane & 15), nHi = nLo + 16;
    const int mbase = m0 + mw + ((lane < 16) ? 0 : 8);
    float* ab = attn + (size_t)b * S_ * S_;
    const float* ef = event_flag + (size_t)b * S_;
#pragma unroll
    for (int r = 0; r < 8; ++r) {
        const int q0 = mbase + r, q1 = mbase + 16 + r;
        const float e0 = ew * ef[q0], e1 = ew * ef[q1];
        ab[(size_t)q0 * S_ + nLo] = c00[r] * scale + e0;
        ab[(size_t)q0 * S_ + nHi] = c01[r] * scale + e0;
        ab[(size_t)q1 * S_ + nLo] = c10[r] * scale + e1;
        ab[(size_t)q1 * S_ + nHi] = c11[r] * scale + e1;
    }
}

// =====================================================================
// Kernel 3: row softmax over 2048 keys, in place. One 256-thread WG/row.
// =====================================================================
__global__ __launch_bounds__(256) void softmax_kernel(float* __restrict__ attn) {
    __shared__ float red[256];
    const int tid = threadIdx.x;
    float* p = attn + (size_t)blockIdx.x * S_;
    float v[8];
    float vmax = -3.4e38f;
#pragma unroll
    for (int j = 0; j < 8; ++j) {
        v[j] = p[tid + 256 * j];
        vmax = fmaxf(vmax, v[j]);
    }
    red[tid] = vmax;
    __syncthreads();
    for (int s = 128; s > 0; s >>= 1) {
        if (tid < s) red[tid] = fmaxf(red[tid], red[tid + s]);
        __syncthreads();
    }
    const float rmax = red[0];
    __syncthreads();
    float sum = 0.f;
#pragma unroll
    for (int j = 0; j < 8; ++j) {
        v[j] = __expf(v[j] - rmax);
        sum += v[j];
    }
    red[tid] = sum;
    __syncthreads();
    for (int s = 128; s > 0; s >>= 1) {
        if (tid < s) red[tid] += red[tid + s];
        __syncthreads();
    }
    const float inv = 1.0f / red[0];
#pragma unroll
    for (int j = 0; j < 8; ++j) p[tid + 256 * j] = v[j] * inv;
}

// =====================================================================
// Kernel 4: context[b,q,h] = sum_k attn[b,q,k] * V[b,k,h]
// WG tile 128(q) x 64(h); attn f32->bf16 on stage-in; V transposed into
// LDS [n][k]; double-buffered, 1 barrier/iter.
// =====================================================================
__global__ __launch_bounds__(256) void context_kernel(const float* __restrict__ attn,
                                                      const bf16_t* __restrict__ V,
                                                      float* __restrict__ ctx) {
    __shared__ __align__(16) bf16_t Asm[2][128 * 32];
    __shared__ __align__(16) bf16_t Bsm[2][64 * 32];
    const int tid = threadIdx.x, lane = tid & 31, wid = tid >> 5;
    const int b = blockIdx.z;
    const int m0 = blockIdx.x * 128, n0 = blockIdx.y * 64;
    const float* Ab = attn + (size_t)b * S_ * S_;
    const bf16_t* Vb = V + (size_t)b * S_ * H_;
    const int mw = (wid & 3) * 32, nw = (wid >> 2) * 32;
    const int NIT = S_ / 32;  // 64

    v4f aReg[4];
    v8bf vReg;
    const int vk = tid >> 3, vc8 = tid & 7;  // V tile cover: 32 k-rows x 8 chunks
    auto loadG = [&](int kt) {
        const int k0 = kt * 32;
#pragma unroll
        for (int j = 0; j < 4; ++j) {  // attn: 128x32 f32 = 1024 float4
            const int idx = tid + j * 256, row = idx >> 3, c4 = idx & 7;
            aReg[j] = *(const v4f*)(Ab + (size_t)(m0 + row) * S_ + k0 + c4 * 4);
        }
        vReg = *(const v8bf*)(Vb + (size_t)(k0 + vk) * H_ + n0 + vc8 * 8);
    };
    auto storeL = [&](int buf) {
#pragma unroll
        for (int j = 0; j < 4; ++j) {
            const int idx = tid + j * 256, row = idx >> 3, c4 = idx & 7;
            v4bf t;
            t[0] = (bf16_t)aReg[j][0]; t[1] = (bf16_t)aReg[j][1];
            t[2] = (bf16_t)aReg[j][2]; t[3] = (bf16_t)aReg[j][3];
            *(v4bf*)(Asm[buf] + row * 32 + c4 * 4) = t;
        }
#pragma unroll
        for (int j = 0; j < 8; ++j)  // transpose V chunk into Bsm[n][k]
            Bsm[buf][(vc8 * 8 + j) * 32 + vk] = vReg[j];
    };

    v8f c00 = {}, c01 = {}, c10 = {}, c11 = {};
    loadG(0);
    storeL(0);
    for (int kt = 0; kt < NIT; ++kt) {
        if (kt + 1 < NIT) loadG(kt + 1);
        __syncthreads();
        const bf16_t* A_ = Asm[kt & 1];
        const bf16_t* Bm_ = Bsm[kt & 1];
        v16bf a0 = ldsA_frag<32>(A_, lane, mw, 0);
        v16bf a1 = ldsA_frag<32>(A_, lane, mw + 16, 0);
        v16bf b0 = ldsB_frag<32>(Bm_, lane, nw, 0);
        v16bf b1 = ldsB_frag<32>(Bm_, lane, nw + 16, 0);
        c00 = wmma_bf16(a0, b0, c00); c01 = wmma_bf16(a0, b1, c01);
        c10 = wmma_bf16(a1, b0, c10); c11 = wmma_bf16(a1, b1, c11);
        if (kt + 1 < NIT) storeL((kt + 1) & 1);
    }
    const int nLo = n0 + nw + (lane & 15), nHi = nLo + 16;
    const int mbase = m0 + mw + ((lane < 16) ? 0 : 8);
    float* cb = ctx + (size_t)b * S_ * H_;
#pragma unroll
    for (int r = 0; r < 8; ++r) {
        cb[(size_t)(mbase + r) * H_ + nLo]      = c00[r];
        cb[(size_t)(mbase + r) * H_ + nHi]      = c01[r];
        cb[(size_t)(mbase + 16 + r) * H_ + nLo] = c10[r];
        cb[(size_t)(mbase + 16 + r) * H_ + nHi] = c11[r];
    }
}

// =====================================================================
extern "C" void kernel_launch(void* const* d_in, const int* in_sizes, int n_in,
                              void* d_out, int out_size, void* d_ws, size_t ws_size,
                              hipStream_t stream) {
    const float* X          = (const float*)d_in[0];
    const float* event_flag = (const float*)d_in[1];
    // d_in[2] = lengths : unused by the reference (no masking)
    const float* Wq = (const float*)d_in[3];
    const float* bq = (const float*)d_in[4];
    const float* Wk = (const float*)d_in[5];
    const float* bk = (const float*)d_in[6];
    const float* Wv = (const float*)d_in[7];
    const float* bv = (const float*)d_in[8];
    const float* ew = (const float*)d_in[9];
    const float* sr = (const float*)d_in[10];

    // Workspace: bf16 Q, K, V (3 * B*S*H * 2B = ~50 MB)
    bf16_t* Qb = (bf16_t*)d_ws;
    bf16_t* Kb = Qb + (size_t)B_ * S_ * H_;
    bf16_t* Vb = Kb + (size_t)B_ * S_ * H_;

    // Outputs: context [B,S,H] then attn [B,S,S], concatenated flat
    float* ctx  = (float*)d_out;
    float* attn = ctx + (size_t)B_ * S_ * H_;

    const dim3 blk(256);
    const dim3 gq((B_ * S_) / 128, H_ / 64);  // 128 x 8
    qkv_kernel<<<gq, blk, 0, stream>>>(X, Wq, bq, Qb);
    qkv_kernel<<<gq, blk, 0, stream>>>(X, Wk, bk, Kb);
    qkv_kernel<<<gq, blk, 0, stream>>>(X, Wv, bv, Vb);

    scores_kernel<<<dim3(S_ / 128, S_ / 64, B_), blk, 0, stream>>>(Qb, Kb, event_flag, ew, sr, attn);
    softmax_kernel<<<B_ * S_, blk, 0, stream>>>(attn);
    context_kernel<<<dim3(S_ / 128, H_ / 64, B_), blk, 0, stream>>>(attn, Vb, ctx);
}
